// Decoder_14620068675902
// MI455X (gfx1250) — matrix operand order repaired and checked by reading, not compiled, not run on hardware
//
#include <hip/hip_runtime.h>
#include <hip/hip_bf16.h>

typedef __attribute__((ext_vector_type(16))) __bf16 v16bf;
typedef __attribute__((ext_vector_type(8)))  float  v8f;

#define DD   64
#define DIN  256
#define NH1  128
#define NH2  64
#define BB   16384
#define KK   32
#define NNE  200000
#define NW   8
#define ROWS (BB * (KK + 1))          // 540672
#define TILES (ROWS / 16)             // 33792

__device__ __forceinline__ unsigned short f2bf(float f) {
    union { float f; unsigned u; } v; v.f = f;
    unsigned u = v.u + 0x7fffu + ((v.u >> 16) & 1u);   // round-to-nearest-even
    return (unsigned short)(u >> 16);
}

union FragB { v16bf v; uint4 q[2]; };
union FragF { v8f  v; float f[8]; };

__global__ __launch_bounds__(NW * 32)
void mlp_pairs_kernel(const float* __restrict__ embeds, const int* __restrict__ pos,
                      const int* __restrict__ neg, const float* __restrict__ W1,
                      const float* __restrict__ b1, const float* __restrict__ W2,
                      const float* __restrict__ b2, const float* __restrict__ W3,
                      const float* __restrict__ b3, float* __restrict__ scores)
{
    // Weights pre-swizzled to WMMA B-fragment order (bf16), per-wave X staging.
    __shared__ uint4 w1q[DIN * NH1 / 8];     // 64 KB
    __shared__ uint4 w2q[NH1 * NH2 / 8];     // 16 KB
    __shared__ float b1s[NH1];
    __shared__ float b2s[NH2];
    __shared__ float w3s[NH2];
    __shared__ uint4 xsq[NW][16 * DIN / 8];  // 8 KB per wave

    const int tid = threadIdx.x;

    // ---- one-time per-block weight preload, swizzled into fragment layout ----
    {
        unsigned short* w1s = (unsigned short*)w1q;
        for (int idx = tid; idx < DIN * NH1; idx += NW * 32) {
            int Kg = idx >> 7, n = idx & 127;
            int kb = Kg >> 5, kr = Kg & 31, nb = n >> 4, nc = n & 15;
            int lane = ((kr >> 4) << 4) | nc;   // B frag: lanes 0-15 K=0..15, 16-31 K=16..31
            int e = kr & 15;
            w1s[((((kb << 3) + nb) * 32 + lane) << 4) + e] = f2bf(W1[idx]);
        }
        unsigned short* w2s = (unsigned short*)w2q;
        for (int idx = tid; idx < NH1 * NH2; idx += NW * 32) {
            int Kg = idx >> 6, n = idx & 63;
            int kb = Kg >> 5, kr = Kg & 31, nb = n >> 4, nc = n & 15;
            int lane = ((kr >> 4) << 4) | nc;
            int e = kr & 15;
            w2s[((((kb << 2) + nb) * 32 + lane) << 4) + e] = f2bf(W2[idx]);
        }
        if (tid < NH1) b1s[tid] = b1[tid];
        if (tid < NH2) { b2s[tid] = b2[tid]; w3s[tid] = W3[tid]; }
    }
    __syncthreads();

    const int w = tid >> 5, lane = tid & 31;
    const int m = lane & 15, h = lane >> 4;   // row in tile, K-half
    const int tile = blockIdx.x * NW + w;
    const int r = tile * 16 + m;

    int i0, i1;
    if (r < BB) { i0 = pos[2 * r];        i1 = pos[2 * r + 1]; }
    else        { int rr = r - BB; i0 = neg[2 * rr]; i1 = neg[2 * rr + 1]; }

    // ---- build 16x256 bf16 feature tile in LDS: f = i*128 + j*64 + c ----
    const float* e0a = embeds + (size_t)i0 * DD;
    const float* e1a = embeds + (size_t)NNE * DD + (size_t)i0 * DD;
    const float* e0b = embeds + (size_t)i1 * DD;
    const float* e1b = embeds + (size_t)NNE * DD + (size_t)i1 * DD;

    #pragma unroll
    for (int c8 = 0; c8 < 4; ++c8) {
        const int c = h * 32 + c8 * 8;   // this lane covers channels [h*32, h*32+32)
        float A0[8], A1[8], B0[8], B1v[8];
        *(float4*)&A0[0]  = *(const float4*)(e0a + c); *(float4*)&A0[4]  = *(const float4*)(e0a + c + 4);
        *(float4*)&A1[0]  = *(const float4*)(e1a + c); *(float4*)&A1[4]  = *(const float4*)(e1a + c + 4);
        *(float4*)&B0[0]  = *(const float4*)(e0b + c); *(float4*)&B0[4]  = *(const float4*)(e0b + c + 4);
        *(float4*)&B1v[0] = *(const float4*)(e1b + c); *(float4*)&B1v[4] = *(const float4*)(e1b + c + 4);

        const float* xa[4] = { A0, A0, A1, A1 };
        const float* xb[4] = { B0, B1v, B0, B1v };
        #pragma unroll
        for (int cb = 0; cb < 4; ++cb) {
            const float* x = xa[cb]; const float* y = xb[cb];
            uint4 o;
            o.x = (unsigned)f2bf(x[0]*y[0]) | ((unsigned)f2bf(x[1]*y[1]) << 16);
            o.y = (unsigned)f2bf(x[2]*y[2]) | ((unsigned)f2bf(x[3]*y[3]) << 16);
            o.z = (unsigned)f2bf(x[4]*y[4]) | ((unsigned)f2bf(x[5]*y[5]) << 16);
            o.w = (unsigned)f2bf(x[6]*y[6]) | ((unsigned)f2bf(x[7]*y[7]) << 16);
            xsq[w][(m * DIN + cb * 64 + c) >> 3] = o;
        }
    }
    asm volatile("s_wait_dscnt 0" ::: "memory");

    // ---- layer 1: X[16x256] x W1[256x128], bf16 WMMA, f32 acc ----
    // A frag (contiguous, matches B swizzle): e=0..15 -> K=16h+e
    FragB a1f[8];
    #pragma unroll
    for (int kb = 0; kb < 8; ++kb) {
        a1f[kb].q[0] = xsq[w][m * 32 + kb * 4 + 2 * h];
        a1f[kb].q[1] = xsq[w][m * 32 + kb * 4 + 2 * h + 1];
    }
    unsigned short* X2 = (unsigned short*)xsq[w];   // staging buffer is dead now; reuse for H1
    #pragma unroll
    for (int nb = 0; nb < 8; ++nb) {
        FragF acc;
        #pragma unroll
        for (int i = 0; i < 8; ++i) acc.f[i] = 0.0f;
        #pragma unroll
        for (int kb = 0; kb < 8; ++kb) {
            FragB bf;
            const int base = (((kb << 3) + nb) * 32 + lane) * 2;
            bf.q[0] = w1q[base]; bf.q[1] = w1q[base + 1];
            acc.v = __builtin_amdgcn_wmma_f32_16x16x32_bf16(
                false, a1f[kb].v, false, bf.v, (short)0, acc.v, false, false);
        }
        const int col = nb * 16 + m;        // C layout: VGPR v -> M=v+8h, N=lane&15
        const float bias = b1s[col];
        #pragma unroll
        for (int v = 0; v < 8; ++v) {
            float hv = acc.f[v] + bias;
            hv = hv > 0.0f ? hv : 0.0f;
            X2[(v + 8 * h) * NH1 + col] = f2bf(hv);
        }
    }
    asm volatile("s_wait_dscnt 0" ::: "memory");

    // ---- layer 2: H1[16x128] x W2[128x64] ----
    FragB a2f[4];
    #pragma unroll
    for (int kb = 0; kb < 4; ++kb) {
        a2f[kb].q[0] = xsq[w][m * 16 + kb * 4 + 2 * h];
        a2f[kb].q[1] = xsq[w][m * 16 + kb * 4 + 2 * h + 1];
    }
    float* H2s = (float*)xsq[w];            // reuse again (4 KB f32 over dead 4 KB bf16)
    #pragma unroll
    for (int nb = 0; nb < 4; ++nb) {
        FragF acc;
        #pragma unroll
        for (int i = 0; i < 8; ++i) acc.f[i] = 0.0f;
        #pragma unroll
        for (int kb = 0; kb < 4; ++kb) {
            FragB bf;
            const int base = (((kb << 2) + nb) * 32 + lane) * 2;
            bf.q[0] = w2q[base]; bf.q[1] = w2q[base + 1];
            acc.v = __builtin_amdgcn_wmma_f32_16x16x32_bf16(
                false, a2f[kb].v, false, bf.v, (short)0, acc.v, false, false);
        }
        const int col = nb * 16 + m;
        const float bias = b2s[col];
        #pragma unroll
        for (int v = 0; v < 8; ++v) {
            float hv = acc.f[v] + bias;
            hv = hv > 0.0f ? hv : 0.0f;
            H2s[(v + 8 * h) * NH2 + col] = hv;
        }
    }
    asm volatile("s_wait_dscnt 0" ::: "memory");

    // ---- layer 3: dot(H2_row, w3) + b3 -> sigmoid -> exp ----
    if (lane < 16) {
        float s = b3[0];
        #pragma unroll 8
        for (int c = 0; c < NH2; ++c) s += H2s[lane * NH2 + c] * w3s[c];
        float sig = 1.0f / (1.0f + __expf(-s));
        scores[tile * 16 + lane] = __expf(sig);
    }
}

__global__ __launch_bounds__(256)
void reduce_b_kernel(const float* __restrict__ scores, float* __restrict__ partials)
{
    __shared__ float sbuf[256];
    const int b = blockIdx.x * 256 + threadIdx.x;
    float term = 0.0f;
    if (b < BB) {
        const float p = scores[b];
        const float* ng = scores + BB + (size_t)b * KK;
        float dn = p;
        #pragma unroll 8
        for (int k = 0; k < KK; ++k) dn += ng[k];
        term = p / (dn + 1e-8f) + 1e-8f;
    }
    sbuf[threadIdx.x] = term;
    __syncthreads();
    for (int s = 128; s > 0; s >>= 1) {
        if (threadIdx.x < s) sbuf[threadIdx.x] += sbuf[threadIdx.x + s];
        __syncthreads();
    }
    if (threadIdx.x == 0) partials[blockIdx.x] = sbuf[0];
}

__global__ __launch_bounds__(256)
void final_kernel(const float* __restrict__ partials, int n, float* __restrict__ out)
{
    __shared__ float sbuf[256];
    float s = 0.0f;
    for (int i = threadIdx.x; i < n; i += 256) s += partials[i];
    sbuf[threadIdx.x] = s;
    __syncthreads();
    for (int st = 128; st > 0; st >>= 1) {
        if (threadIdx.x < st) sbuf[threadIdx.x] += sbuf[threadIdx.x + st];
        __syncthreads();
    }
    if (threadIdx.x == 0) out[0] = -sbuf[0];
}

extern "C" void kernel_launch(void* const* d_in, const int* in_sizes, int n_in,
                              void* d_out, int out_size, void* d_ws, size_t ws_size,
                              hipStream_t stream)
{
    const float* embeds = (const float*)d_in[0];
    const int*   pos    = (const int*)d_in[1];
    const int*   neg    = (const int*)d_in[2];
    const float* W1     = (const float*)d_in[3];
    const float* b1     = (const float*)d_in[4];
    const float* W2     = (const float*)d_in[5];
    const float* b2     = (const float*)d_in[6];
    const float* W3     = (const float*)d_in[7];
    const float* b3     = (const float*)d_in[8];

    float* scores   = (float*)d_ws;                                    // ROWS floats
    float* partials = (float*)((char*)d_ws + (size_t)ROWS * sizeof(float)); // BB/256 floats

    mlp_pairs_kernel<<<TILES / NW, NW * 32, 0, stream>>>(
        embeds, pos, neg, W1, b1, W2, b2, W3, b3, scores);
    reduce_b_kernel<<<BB / 256, 256, 0, stream>>>(scores, partials);
    final_kernel<<<1, 256, 0, stream>>>(partials, BB / 256, (float*)d_out);
}
